// SupConBinaryLoss_83657372991614
// MI455X (gfx1250) — matrix operand order repaired and checked by reading, not compiled, not run on hardware
//
#include <hip/hip_runtime.h>
#include <hip/hip_bf16.h>

typedef __attribute__((ext_vector_type(16))) _Float16 v16h;
typedef __attribute__((ext_vector_type(8)))  _Float16 v8h;
typedef __attribute__((ext_vector_type(8)))  float    v8f;

#define B_N 8192
#define D_K 128
#define TOPK 32
#define SLICES 4                         // waves per block, each covers B_N/16/SLICES M-tiles
#define MT_TOTAL (B_N / 16)              // 512 candidate tiles
#define MT_PER_SLICE (MT_TOTAL / SLICES) // 128
#define SENT (-1.0e30f)                  // finite -inf sentinel (branchless-safe)
#define LN2F 0.6931471805599453f
// sqrt(INV_TAU * log2(e)) = sqrt(5 * 1.4426950408889634)
#define PRESCALE 2.68579397f

// ---------------- prep: f32 -> pre-scaled f16 copy of z ----------------
// acc of z16 row-dots lands directly in base-2 logit units: u = (dot/tau)*log2(e)
__global__ void prep_f16(const float* __restrict__ z, _Float16* __restrict__ z16, int n) {
    int i = blockIdx.x * blockDim.x + threadIdx.x;
    if (i < n) z16[i] = (_Float16)(z[i] * PRESCALE);
}

// A-fragment: 16x32 f16 tile slice per lane, 4 chunks for K=128
struct AFrag { v16h c[4]; };

__device__ __forceinline__ void load_a(const _Float16* __restrict__ z16,
                                       int row, int half, AFrag& f) {
    const _Float16* ap = z16 + (size_t)row * D_K + half * 16;
#pragma unroll
    for (int kc = 0; kc < 4; ++kc) {
        // contiguous K layout (matches B): lane half 0 -> K {0..15}, half 1 -> {16..31}
        v8h lo8 = *(const v8h*)(ap + kc * 32);
        v8h hi8 = *(const v8h*)(ap + kc * 32 + 8);
#pragma unroll
        for (int e = 0; e < 8; ++e) { f.c[kc][e] = lo8[e]; f.c[kc][e + 8] = hi8[e]; }
    }
}

// WMMA + single self-free epilogue (self handled by scalar-branch acc patch)
__device__ __forceinline__ void process_tile(
    int mt, const AFrag& f, const v16h (&bf)[4],
    const unsigned* sbits, float* myk,
    unsigned labxor, unsigned smask_self, int selftile, int half, int col,
    float& m, float& s_all, float& s_pos, float& possum,
    int& poscnt, int& negcnt, float& cmin, int& cidx)
{
    v8f acc = {};
#pragma unroll
    for (int kc = 0; kc < 4; ++kc)
        acc = __builtin_amdgcn_wmma_f32_16x16x32_f16(
            false, f.c[kc], false, bf[kc], (short)0, acc, false, false);

    const int jbase = mt * 16 + half * 8;
    unsigned diffb = ((sbits[jbase >> 5] >> (jbase & 31)) & 0xffu) ^ labxor;

    if (__builtin_expect(mt == selftile, 0)) {   // lane-uniform scalar branch
#pragma unroll
        for (int r = 0; r < 8; ++r)
            acc[r] = ((smask_self >> r) & 1u) ? SENT : acc[r];
        diffb |= smask_self;                     // self counts as negative (SENT value)
    }

    const unsigned negm = diffb & 0xffu;
    const unsigned posm = (~diffb) & 0xffu;

    float tmax = SENT;                            // plain tile max (self already SENT)
#pragma unroll
    for (int r = 0; r < 8; ++r) tmax = fmaxf(tmax, acc[r]);
    if (tmax > m) {                               // at most once per tile
        const float rs = __builtin_amdgcn_exp2f(m - tmax);
        s_all *= rs; s_pos *= rs; m = tmax;
    }
#pragma unroll
    for (int r = 0; r < 8; ++r) {
        const float u = acc[r];
        const float e = __builtin_amdgcn_exp2f(u - m);   // SENT -> 0
        const float fp = (float)((posm >> r) & 1u);
        s_all  += e;                                     // self contributes 0
        s_pos   = fmaf(fp, e, s_pos);
        possum  = fmaf(fp, u, possum);
    }
    poscnt += __popc(posm);
    negcnt += __popc(negm);

    if (tmax > cmin) {                            // rare top-32 maintenance path
#pragma unroll
        for (int r = 0; r < 8; ++r) {
            const float w = ((negm >> r) & 1u) ? acc[r] : SENT;
            if (w > cmin) {
                myk[cidx] = w;
                cmin = myk[0]; cidx = 0;
#pragma unroll
                for (int k = 1; k < TOPK; ++k) {
                    const float x = myk[k];
                    if (x < cmin) { cmin = x; cidx = k; }
                }
            }
        }
    }
}

// ---------------- main fused kernel ----------------
// Block = 16 anchors (WMMA N dim). 4 waves slice the 512 candidate tiles (M dim).
// K=128 via 4x v_wmma_f32_16x16x32_f16 with resident B fragments.
__global__ __launch_bounds__(32 * SLICES) void supcon_main(
    const _Float16* __restrict__ z16, const int* __restrict__ labels,
    float* __restrict__ partials)
{
    __shared__ unsigned s_bits[B_N / 32];                 // 1 KB label bitmap
    __shared__ float s_topk[SLICES][32][TOPK + 1];        // ~16.5 KB, bank-padded
    __shared__ float s_m[SLICES][16], s_sa[SLICES][16], s_sp[SLICES][16];
    __shared__ float s_ps[SLICES][16], s_pc[SLICES][16], s_nc[SLICES][16];

    const int tid  = threadIdx.x;
    const int lane = tid & 31;
    const int wv   = tid >> 5;                  // slice id 0..3
    const int col  = lane & 15;                 // anchor slot / tile column N
    const int half = lane >> 4;                 // 0 or 1 (lane half)
    const int ia   = blockIdx.x * 16 + col;     // this lane's anchor index
    const int mstart = wv * MT_PER_SLICE;
    const int selftile = (int)blockIdx.x;       // uniform: the one tile containing j==ia

    // build the label bitmap with ballots (32 labels per dword)
#pragma unroll 1
    for (int i = 0; i < (B_N / 32) / SLICES; ++i) {
        const int dw = wv * ((B_N / 32) / SLICES) + i;
        const int lab = labels[dw * 32 + lane];
        const unsigned long long b = __ballot(lab & 1);
        if (lane == 0) s_bits[dw] = (unsigned)b;
    }

    float* myk = &s_topk[wv][lane][0];
#pragma unroll
    for (int k = 0; k < TOPK; ++k) myk[k] = SENT;
    float cmin = SENT; int cidx = 0;

    __syncthreads();

    const unsigned labxor = ((s_bits[ia >> 5] >> (ia & 31)) & 1u) ? 0xffu : 0u;
    // self bit inside the self tile (only lanes whose half holds row ia&15)
    const unsigned smask_self = (half == (col >> 3)) ? (1u << (col & 7)) : 0u;

    // resident B fragments (anchor columns), contiguous K layout (matches A): half 0 -> K {0..15}, half 1 -> {16..31}
    v16h bfrag[4];
    {
        const _Float16* bp = z16 + (size_t)ia * D_K + half * 16;
#pragma unroll
        for (int kc = 0; kc < 4; ++kc) bfrag[kc] = *(const v16h*)(bp + kc * 32);
    }

    // single running max (base-2 domain); s_all and s_pos share it
    float m = SENT, s_all = 0.f, s_pos = 0.f, possum = 0.f;
    int poscnt = 0, negcnt = 0;

    // software-pipelined M loop: load tile t+1 while computing tile t
    AFrag fa, fb;
    load_a(z16, mstart * 16 + col, half, fa);
    for (int t = 0; t < MT_PER_SLICE; t += 2) {
        const int mt0 = mstart + t;
        const int mt1 = mstart + t + 1;
        const int mt2 = (t + 2 < MT_PER_SLICE) ? mstart + t + 2 : mstart;
        load_a(z16, mt1 * 16 + col, half, fb);
        process_tile(mt0, fa, bfrag, s_bits, myk, labxor, smask_self, selftile, half, col,
                     m, s_all, s_pos, possum, poscnt, negcnt, cmin, cidx);
        load_a(z16, mt2 * 16 + col, half, fa);
        process_tile(mt1, fb, bfrag, s_bits, myk, labxor, smask_self, selftile, half, col,
                     m, s_all, s_pos, possum, poscnt, negcnt, cmin, cidx);
    }

    // undo the self-as-negative count (once, in the wave owning the self tile)
    if (selftile >= mstart && selftile < mstart + MT_PER_SLICE && smask_self)
        negcnt -= 1;

    // merge the two half-lanes of each anchor (branchless, finite sentinels)
    {
        const float m2  = __shfl_xor(m, 16);
        const float sa2 = __shfl_xor(s_all, 16);
        const float sp2 = __shfl_xor(s_pos, 16);
        const float M   = fmaxf(m, m2);
        const float ea  = __builtin_amdgcn_exp2f(m - M);
        const float eb  = __builtin_amdgcn_exp2f(m2 - M);
        s_all = s_all * ea + sa2 * eb;
        s_pos = s_pos * ea + sp2 * eb;
        m = M;
        possum += __shfl_xor(possum, 16);
        poscnt += __shfl_xor(poscnt, 16);
        negcnt += __shfl_xor(negcnt, 16);
    }
    if (lane < 16) {
        s_m[wv][lane] = m;   s_sa[wv][lane] = s_all; s_sp[wv][lane] = s_pos;
        s_ps[wv][lane] = possum; s_pc[wv][lane] = (float)poscnt; s_nc[wv][lane] = (float)negcnt;
    }
    __syncthreads();

    // wave 0 merges the 4 slices and finishes per-anchor math
    float pf = 0.f, cf = 0.f, pm = 0.f, cm = 0.f;
    if (wv == 0 && lane < 16) {
        float M = SENT;
#pragma unroll
        for (int s = 0; s < SLICES; ++s) M = fmaxf(M, s_m[s][lane]);
        float SA = 0.f, SP = 0.f, PS = 0.f; int PC = 0, NC = 0;
#pragma unroll
        for (int s = 0; s < SLICES; ++s) {
            const float e = __builtin_amdgcn_exp2f(s_m[s][lane] - M);
            SA += s_sa[s][lane] * e;
            SP += s_sp[s][lane] * e;
            PS += s_ps[s][lane];
            PC += (int)s_pc[s][lane];
            NC += (int)s_nc[s][lane];
        }

        // top-32 of 256 merged negative candidates (destructive max-selection)
        float mt_ = SENT, st_ = 0.f;
        for (int t = 0; t < TOPK; ++t) {
            float best = SENT; int bq = 0;
            for (int q = 0; q < SLICES * 2 * TOPK; ++q) {
                const float x = s_topk[q >> 6][lane + ((q >> 5) & 1) * 16][q & 31];
                if (x > best) { best = x; bq = q; }
            }
            if (best <= -1.0e29f) break;         // only sentinels left
            if (best > mt_) { st_ = st_ * __builtin_amdgcn_exp2f(mt_ - best) + 1.f; mt_ = best; }
            else             st_ += __builtin_amdgcn_exp2f(best - mt_);
            s_topk[bq >> 6][lane + ((bq >> 5) & 1) * 16][bq & 31] = SENT;
        }

        // lse_mined = logaddexp(lse_pos, lse_topk), still base-2
        const float MM = fmaxf(M, mt_);
        const float SM = SP * __builtin_amdgcn_exp2f(M - MM)
                       + st_ * __builtin_amdgcn_exp2f(mt_ - MM);

        const float meanpos_u = PS / fmaxf((float)PC, 1.f);
        const bool vf = PC > 0;
        const bool vm = vf && (NC > 0);
        // convert to natural log units with a single *ln2
        const float lf = ((M  + __builtin_amdgcn_logf(SA)) - meanpos_u) * LN2F;
        const float lm = ((MM + __builtin_amdgcn_logf(SM)) - meanpos_u) * LN2F;
        pf = vf ? lf : 0.f; cf = vf ? 1.f : 0.f;
        pm = vm ? lm : 0.f; cm = vm ? 1.f : 0.f;
    }
    if (wv == 0) {
#pragma unroll
        for (int off = 16; off >= 1; off >>= 1) {
            pf += __shfl_xor(pf, off);
            cf += __shfl_xor(cf, off);
            pm += __shfl_xor(pm, off);
            cm += __shfl_xor(cm, off);
        }
        if (lane == 0) ((float4*)partials)[blockIdx.x] = make_float4(pf, cf, pm, cm);
    }
}

// ---------------- deterministic final reduction ----------------
__global__ void supcon_final(const float* __restrict__ partials,
                             const float* __restrict__ alpha_p,
                             float* __restrict__ out, int nrec)
{
    __shared__ float4 red[256];
    int tid = threadIdx.x;
    float4 a = make_float4(0.f, 0.f, 0.f, 0.f);
    const float4* p = (const float4*)partials;
    for (int r = tid; r < nrec; r += 256) {
        float4 v = p[r];
        a.x += v.x; a.y += v.y; a.z += v.z; a.w += v.w;
    }
    red[tid] = a;
    __syncthreads();
    for (int s = 128; s >= 1; s >>= 1) {
        if (tid < s) {
            float4 b = red[tid + s];
            red[tid].x += b.x; red[tid].y += b.y; red[tid].z += b.z; red[tid].w += b.w;
        }
        __syncthreads();
    }
    if (tid == 0) {
        float4 t = red[0];
        float sum_lf = t.x, cfv = t.y, sum_lm = t.z, cmv = t.w;
        float alpha = alpha_p[0];
        float loss_full      = sum_lf / fmaxf(cfv, 1.f);
        float loss_mined_raw = sum_lm / fmaxf(cmv, 1.f);
        float loss_mined = (cmv > 0.f) ? loss_mined_raw : loss_full;
        float loss = (cfv > 0.f) ? ((1.f - alpha) * loss_full + alpha * loss_mined) : 0.f;
        out[0] = loss;
    }
}

extern "C" void kernel_launch(void* const* d_in, const int* in_sizes, int n_in,
                              void* d_out, int out_size, void* d_ws, size_t ws_size,
                              hipStream_t stream) {
    const float* z      = (const float*)d_in[0];
    const int*   labels = (const int*)d_in[1];
    // d_in[2] = topk_neg (always 32 here; compiled-in as TOPK)
    const float* alpha  = (const float*)d_in[3];
    float* out = (float*)d_out;

    _Float16* z16 = (_Float16*)d_ws;
    float* partials = (float*)((char*)d_ws + (size_t)B_N * D_K * sizeof(_Float16));

    int n = B_N * D_K;
    prep_f16<<<(n + 255) / 256, 256, 0, stream>>>(z, z16, n);

    int nblocks = B_N / 16;   // 512 blocks, 4 waves each
    supcon_main<<<nblocks, 32 * SLICES, 0, stream>>>(z16, labels, partials);
    supcon_final<<<1, 256, 0, stream>>>(partials, alpha, out, nblocks);
}
